// DSAttention_5196910428223
// MI455X (gfx1250) — matrix operand order repaired and checked by reading, not compiled, not run on hardware
//
#include <hip/hip_runtime.h>

typedef __attribute__((ext_vector_type(16))) __bf16       v16bf;
typedef __attribute__((ext_vector_type(4)))  __bf16       v4bf;
typedef __attribute__((ext_vector_type(2)))  __bf16       v2bf;
typedef __attribute__((ext_vector_type(8)))  float        v8f;
typedef __attribute__((ext_vector_type(4)))  unsigned int v4u;
typedef __attribute__((ext_vector_type(8)))  unsigned int v8u;

// Pointee type the async-LDS builtin expects (from hipcc diagnostic).
typedef int vsi4 __attribute__((__vector_size__(4 * sizeof(int))));
typedef __attribute__((address_space(1))) vsi4* gp_v4i;   // global
typedef __attribute__((address_space(3))) vsi4* lp_v4i;   // LDS

#define WMMA_BF16(a, b, c) \
  __builtin_amdgcn_wmma_f32_16x16x32_bf16(false, (a), false, (b), (short)0, (c), false, false)

#if __has_builtin(__builtin_amdgcn_global_load_async_to_lds_b128)
#define HAVE_ASYNC_LDS 1
#else
#define HAVE_ASYNC_LDS 0
#endif

constexpr int Bb = 4, Ll = 2048, Ss = 2048, Hh = 8, Dd = 64;
constexpr int ROWS_PER_BLOCK = 128;   // 8 waves x 16 query rows
constexpr int CHUNK   = 64;           // S-chunk per iteration (4 s-tiles)
constexpr int NCHUNK  = Ss / CHUNK;
constexpr int KSTRIDE = 72;           // bf16 units; 144B rows, 16B-aligned chunks
constexpr int VSTRIDE = 72;           // bf16 units; V transposed [d][s]

__device__ __forceinline__ v16bf ld2x16(const __bf16* p0, const __bf16* p1) {
  v4u a = *(const v4u*)p0;
  v4u b = *(const v4u*)p1;
  v8u r;
  r[0] = a[0]; r[1] = a[1]; r[2] = a[2]; r[3] = a[3];
  r[4] = b[0]; r[5] = b[1]; r[6] = b[2]; r[7] = b[3];
  return __builtin_bit_cast(v16bf, r);
}

__device__ __forceinline__ unsigned int pack_bf16(float a, float b) {
  v2bf t; t[0] = (__bf16)a; t[1] = (__bf16)b;
  return __builtin_bit_cast(unsigned int, t);
}

// Copy one 16B segment global->LDS (async DMA when available).
__device__ __forceinline__ void cp16(__bf16* dst, const __bf16* src) {
#if HAVE_ASYNC_LDS
  __builtin_amdgcn_global_load_async_to_lds_b128(
      (gp_v4i)(src), (lp_v4i)(dst), 0, 0);
#else
  *(v4u*)dst = *(const v4u*)src;
#endif
}

__device__ __forceinline__ void wait_async_stage() {
#if HAVE_ASYNC_LDS
#if __has_builtin(__builtin_amdgcn_s_wait_asynccnt)
  __builtin_amdgcn_s_wait_asynccnt(0);
#else
  asm volatile("s_wait_asynccnt 0x0" ::: "memory");
#endif
#endif
}

// Stage one 64x64 bf16 K chunk + 64x64 V' chunk into LDS buffers.
__device__ __forceinline__ void stage_chunk(__bf16* Kdst, __bf16* Vdst,
                                            const __bf16* kbp, const __bf16* vbp,
                                            int tid) {
#pragma unroll
  for (int rr = 0; rr < 2; ++rr) {
    const int idx = tid + rr * 256;       // 0..511
    const int row = idx >> 3;             // 0..63
    const int c8  = (idx & 7) << 3;       // 0..56 (8 bf16 = 16B)
    cp16(Kdst + row * KSTRIDE + c8, kbp + (size_t)row * Dd + c8);
    cp16(Vdst + row * VSTRIDE + c8, vbp + (size_t)row * Ss + c8);
  }
}

// ---------------- Pass 1: one-time K/V fp32 -> bf16 convert (+ V transpose) ----------------
__global__ __launch_bounds__(256) void convert_kv(
    const float* __restrict__ k, const float* __restrict__ v,
    __bf16* __restrict__ Kw, __bf16* __restrict__ Vw) {
  __shared__ __bf16 T[64 * 72];

  const int tid   = threadIdx.x;
  const int stile = blockIdx.x % (Ss / 64);
  const int bh    = blockIdx.x / (Ss / 64);
  const int h     = bh % Hh;
  const int b     = bh / Hh;
  const int s0    = stile * 64;

  const float* kp = k + (((size_t)b * Ss + s0) * Hh + h) * Dd;
  const float* vp = v + (((size_t)b * Ss + s0) * Hh + h) * Dd;
  __bf16* kout = Kw + ((size_t)bh * Ss + s0) * Dd;

#pragma unroll
  for (int rr = 0; rr < 4; ++rr) {
    const int idx  = tid + rr * 256;
    const int srow = idx >> 4;
    const int d4   = (idx & 15) << 2;
    const float4 kf = *(const float4*)(kp + (size_t)srow * (Hh * Dd) + d4);
    v4bf kb;
    kb[0] = (__bf16)kf.x; kb[1] = (__bf16)kf.y;
    kb[2] = (__bf16)kf.z; kb[3] = (__bf16)kf.w;
    *(v4bf*)(kout + srow * Dd + d4) = kb;
    const float4 vf = *(const float4*)(vp + (size_t)srow * (Hh * Dd) + d4);
    T[(d4 + 0) * 72 + srow] = (__bf16)vf.x;
    T[(d4 + 1) * 72 + srow] = (__bf16)vf.y;
    T[(d4 + 2) * 72 + srow] = (__bf16)vf.z;
    T[(d4 + 3) * 72 + srow] = (__bf16)vf.w;
  }
  __syncthreads();
  __bf16* vout = Vw + (size_t)bh * Dd * Ss + s0;
#pragma unroll
  for (int rr = 0; rr < 2; ++rr) {
    const int idx  = tid + rr * 256;
    const int drow = idx >> 3;
    const int c8   = (idx & 7) << 3;
    const v4u t = *(const v4u*)(T + drow * 72 + c8);
    *(v4u*)(vout + (size_t)drow * Ss + c8) = t;
  }
}

// ---------------- Pass 2: transposed flash attention, double-buffered async staging ----------------
__global__ __launch_bounds__(256, 1) void dsattn_flash_wmma_t(
    const float* __restrict__ q, const __bf16* __restrict__ Kw,
    const __bf16* __restrict__ Vw, const float* __restrict__ tau,
    const float* __restrict__ delta, float* __restrict__ out) {
  __shared__ __bf16 Ksh[2][CHUNK * KSTRIDE];   // double-buffered K chunk [s][d]
  __shared__ __bf16 Vsh[2][Dd * VSTRIDE];      // double-buffered V' chunk [d][s]
  __shared__ float  Dall[Ss];                  // entire delta row for this batch (8 KB)

  const int tid  = threadIdx.x;
  const int wave = tid >> 5;
  const int lane = tid & 31;
  const int col  = lane & 15;
  const int half = lane >> 4;

  const int blocksPerSeq = Ll / ROWS_PER_BLOCK;
  const int qblock = blockIdx.x % blocksPerSeq;
  const int bh     = blockIdx.x / blocksPerSeq;
  const int h      = bh % Hh;
  const int b      = bh / Hh;
  const int qbase  = qblock * ROWS_PER_BLOCK + wave * 16;

  const float tau_b = tau[b];
  const float cexp  = 0.125f * 1.4426950408889634f;   // (1/sqrt(64)) * log2(e)

  // ---- Q^T as B-operands: lane = q column, VGPR j holds d = 32*sl + 16*half + 2j ----
  v16bf Bq[2];
  {
    const float* qp = q + (((size_t)b * Ll + (qbase + col)) * Hh + h) * Dd;
#pragma unroll
    for (int sl = 0; sl < 2; ++sl) {
      v16bf bq;
#pragma unroll
      for (int j = 0; j < 8; ++j) {
        const int d = sl * 32 + half * 16 + 2 * j;
        const float2 f = *(const float2*)(qp + d);
        bq[2 * j]     = (__bf16)f.x;
        bq[2 * j + 1] = (__bf16)f.y;
      }
      Bq[sl] = bq;
    }
  }

  v16bf onesA;
#pragma unroll
  for (int i = 0; i < 16; ++i) onesA[i] = (__bf16)1.0f;

  v8f   O[4];
  v8f   Lacc;
  float m_run = -__builtin_inff();
#pragma unroll
  for (int t = 0; t < 4; ++t) O[t] = (v8f){};
  Lacc = (v8f){};

  const __bf16* kb0 = Kw + (size_t)bh * Ss * Dd;   // [s][64]
  const __bf16* vb0 = Vw + (size_t)bh * Dd * Ss;   // [d][S]

  // ---- Prologue: whole delta row to LDS; stage chunk 0 into buffer 0 ----
  {
    const float* dp = delta + (size_t)b * Ss;
#pragma unroll
    for (int rr = 0; rr < 2; ++rr) {
      const int idx = tid + rr * 256;              // 0..511 float4s
      *(float4*)(Dall + idx * 4) = *(const float4*)(dp + idx * 4);
    }
  }
  stage_chunk(Ksh[0], Vsh[0], kb0, vb0, tid);
  wait_async_stage();
  __syncthreads();

  for (int it = 0; it < NCHUNK; ++it) {
    const int cur   = it & 1;
    const int chunk = it * CHUNK;

    // Issue next chunk's staging into the alternate buffer (async DMA overlaps compute).
    if (it + 1 < NCHUNK) {
      const int nc = chunk + CHUNK;
      stage_chunk(Ksh[1 - cur], Vsh[1 - cur],
                  kb0 + (size_t)nc * Dd, vb0 + nc, tid);
    }

    const __bf16* Kb = Ksh[cur];
    const __bf16* Vb = Vsh[cur];
    const float*  Dc = Dall + chunk;

    // ---- S^T tiles: st[t] (16s x 16q) = K_tile @ Q^T ----
    v8f st[4];
#pragma unroll
    for (int t = 0; t < 4; ++t) {
      v8f acc = (v8f){};
#pragma unroll
      for (int ds = 0; ds < 2; ++ds) {
        const __bf16* ap = Kb + (t * 16 + col) * KSTRIDE + ds * 32 + half * 16;
        acc = WMMA_BF16(ld2x16(ap, ap + 8), Bq[ds], acc);
      }
      st[t] = acc;
    }

    // ---- x = tau*s + delta; per-lane in-register max ----
    float xv[4][8];
    float mx = -__builtin_inff();
#pragma unroll
    for (int t = 0; t < 4; ++t) {
      const float4 d0 = *(const float4*)(Dc + 16 * t + 8 * half);
      const float4 d1 = *(const float4*)(Dc + 16 * t + 8 * half + 4);
      xv[t][0] = st[t][0] * tau_b + d0.x;  xv[t][1] = st[t][1] * tau_b + d0.y;
      xv[t][2] = st[t][2] * tau_b + d0.z;  xv[t][3] = st[t][3] * tau_b + d0.w;
      xv[t][4] = st[t][4] * tau_b + d1.x;  xv[t][5] = st[t][5] * tau_b + d1.y;
      xv[t][6] = st[t][6] * tau_b + d1.z;  xv[t][7] = st[t][7] * tau_b + d1.w;
      const float a = fmaxf(fmaxf(xv[t][0], xv[t][1]), fmaxf(xv[t][2], xv[t][3]));
      const float c = fmaxf(fmaxf(xv[t][4], xv[t][5]), fmaxf(xv[t][6], xv[t][7]));
      mx = fmaxf(mx, fmaxf(a, c));
    }
    mx = fmaxf(mx, __shfl_xor(mx, 16, 32));
    const float mnew  = fmaxf(m_run, mx);
    const float alpha = exp2f((m_run - mnew) * cexp);
    m_run = mnew;
#pragma unroll
    for (int i = 0; i < 8; ++i) Lacc[i] *= alpha;
#pragma unroll
    for (int t = 0; t < 4; ++t)
#pragma unroll
      for (int i = 0; i < 8; ++i) O[t][i] *= alpha;

    // ---- exps packed to bf16 pairs ----
    unsigned int u[4][4];
#pragma unroll
    for (int t = 0; t < 4; ++t)
#pragma unroll
      for (int j = 0; j < 4; ++j) {
        const float e0 = exp2f((xv[t][2 * j]     - mnew) * cexp);
        const float e1 = exp2f((xv[t][2 * j + 1] - mnew) * cexp);
        u[t][j] = pack_bf16(e0, e1);
      }

    // ---- P^T B-operands via one cross-half register exchange per slice ----
    v16bf Bp[2];
#pragma unroll
    for (int sl = 0; sl < 2; ++sl) {
      unsigned int recv[4];
#pragma unroll
      for (int j = 0; j < 4; ++j) {
        const unsigned int payload = half ? u[2 * sl][j] : u[2 * sl + 1][j];
        recv[j] = (unsigned int)__shfl_xor((int)payload, 16, 32);
      }
      v8u r;
#pragma unroll
      for (int j = 0; j < 4; ++j) {
        r[j]     = half ? recv[j]          : u[2 * sl][j];
        r[4 + j] = half ? u[2 * sl + 1][j] : recv[j];
      }
      Bp[sl] = __builtin_bit_cast(v16bf, r);
    }

    // ---- Denominator + Out^T ----
#pragma unroll
    for (int sl = 0; sl < 2; ++sl) Lacc = WMMA_BF16(onesA, Bp[sl], Lacc);
#pragma unroll
    for (int dt = 0; dt < 4; ++dt) {
#pragma unroll
      for (int sl = 0; sl < 2; ++sl) {
        const __bf16* ap = Vb + (dt * 16 + col) * VSTRIDE + sl * 32 + half * 16;
        O[dt] = WMMA_BF16(ld2x16(ap, ap + 8), Bp[sl], O[dt]);
      }
    }

    // Next buffer ready + all waves done reading current buffer.
    wait_async_stage();
    __syncthreads();
  }

  // ---- Epilogue ----
  {
    const float inv = 1.0f / Lacc[0];
    float* op = out + (((size_t)b * Ll + (qbase + col)) * Hh + h) * Dd;
#pragma unroll
    for (int dt = 0; dt < 4; ++dt) {
      float4 lo, hi;
      lo.x = O[dt][0] * inv; lo.y = O[dt][1] * inv;
      lo.z = O[dt][2] * inv; lo.w = O[dt][3] * inv;
      hi.x = O[dt][4] * inv; hi.y = O[dt][5] * inv;
      hi.z = O[dt][6] * inv; hi.w = O[dt][7] * inv;
      *(float4*)(op + dt * 16 + half * 8)     = lo;
      *(float4*)(op + dt * 16 + half * 8 + 4) = hi;
    }
  }
}

extern "C" void kernel_launch(void* const* d_in, const int* in_sizes, int n_in,
                              void* d_out, int out_size, void* d_ws, size_t ws_size,
                              hipStream_t stream) {
  const float* q     = (const float*)d_in[0];
  const float* k     = (const float*)d_in[1];
  const float* v     = (const float*)d_in[2];
  const float* tau   = (const float*)d_in[3];
  const float* delta = (const float*)d_in[4];
  float* out = (float*)d_out;
  (void)in_sizes; (void)n_in; (void)out_size; (void)ws_size;

  __bf16* Kw = (__bf16*)d_ws;
  __bf16* Vw = Kw + (size_t)Bb * Hh * Ss * Dd;

  convert_kv<<<dim3(Bb * Hh * (Ss / 64)), dim3(256), 0, stream>>>(k, v, Kw, Vw);

  const dim3 grid(Bb * Hh * (Ll / ROWS_PER_BLOCK));  // 512 blocks
  const dim3 block(256);                             // 8 waves (wave32)
  dsattn_flash_wmma_t<<<grid, block, 0, stream>>>(q, Kw, Vw, tau, delta, out);
}
